// LSTM_23742579212919
// MI455X (gfx1250) — matrix operand order repaired and checked by reading, not compiled, not run on hardware
//
#include <hip/hip_runtime.h>
#include <hip/hip_bf16.h>
#include <hip/hip_fp16.h>

typedef __attribute__((ext_vector_type(16))) _Float16 v16h;
typedef __attribute__((ext_vector_type(8)))  _Float16 v8h;
typedef __attribute__((ext_vector_type(8)))  float    v8f;

// Optional gfx1250 async global->LDS path (guarded; falls back to plain loads)
#if defined(__has_builtin)
#  if __has_builtin(__builtin_amdgcn_global_load_async_to_lds_b128) && \
      __has_builtin(__builtin_amdgcn_s_wait_asynccnt)
#    define USE_ASYNC_LDS 1
#  endif
#endif
#ifndef USE_ASYNC_LDS
#  define USE_ASYNC_LDS 0
#endif

// Problem sizes
static constexpr int kB = 64;      // batch
static constexpr int kT = 512;     // seq len
static constexpr int kE = 256;     // embed
static constexpr int kH = 512;     // hidden
static constexpr int kG = 2048;    // 4*H gates
static constexpr int kNWG = 16;    // persistent WGs in recurrence

// Workspace layout (bytes)
static constexpr size_t WXT_OFF = 0;                       // f16 [2048][256]  (Wx transposed)
static constexpr size_t WHT_OFF = 1u << 20;                // f16 [2048][512]  (Wh transposed)
static constexpr size_t H0_OFF  = 3u << 20;                // f16 [64][512]
static constexpr size_t H1_OFF  = H0_OFF + 64 * 1024;      // f16 [64][512]
static constexpr size_t CNT_OFF = H0_OFF + 128 * 1024;     // int [512] barrier counters
static constexpr size_t XP_OFF  = 4u << 20;                // f16 [T*B][2048] Xproj (incl. bias)

__device__ __forceinline__ v8f zero8() {
  v8f z;
#pragma unroll
  for (int i = 0; i < 8; ++i) z[i] = 0.0f;
  return z;
}
__device__ __forceinline__ v16h cat16(v8h lo, v8h hi) {
  return __builtin_shufflevector(lo, hi, 0,1,2,3,4,5,6,7,8,9,10,11,12,13,14,15);
}
__device__ __forceinline__ float sigf(float x) { return 1.0f / (1.0f + __expf(-x)); }

#if USE_ASYNC_LDS
// Probe-revealed signature: (v4i AS(1)*, v4i AS(3)*, imm offset, imm cpol)
typedef int v4i_async __attribute__((__vector_size__(16)));
typedef __attribute__((address_space(1))) v4i_async* gv4i_p;
typedef __attribute__((address_space(3))) v4i_async* lv4i_p;

__device__ __forceinline__ void async_copy_b128(const _Float16* g, _Float16* l) {
  __builtin_amdgcn_global_load_async_to_lds_b128(
      (gv4i_p)(const_cast<_Float16*>(g)), (lv4i_p)(l), 0, 0);
}
#endif

// ---------------------------------------------------------------------------
// Kernel 0: convert Wg (f32, [768][2048]) into transposed f16 WxT[n][k], WhT[n][k]
// ---------------------------------------------------------------------------
__global__ __launch_bounds__(256) void cvt_weights(const float* __restrict__ Wg,
                                                   _Float16* __restrict__ WxT,
                                                   _Float16* __restrict__ WhT) {
  int idx = blockIdx.x * 256 + threadIdx.x;   // over 768*2048
  int k = idx >> 11;
  int n = idx & 2047;
  _Float16 v = (_Float16)Wg[idx];
  if (k < kE) WxT[(size_t)n * kE + k] = v;
  else        WhT[(size_t)n * kH + (k - kE)] = v;
}

// ---------------------------------------------------------------------------
// Kernel 1: Xproj[t*64+b][n] = emb[x[b,t]] @ Wx + bg   (f16 output, f32 accum)
// Grid: (256 M-tiles of 128 rows, 16 N-tiles of 128 cols). WMMA f16 16x16x32.
// ---------------------------------------------------------------------------
__global__ __launch_bounds__(256) void xproj_kernel(const int* __restrict__ x,
                                                    const float* __restrict__ emb,
                                                    const _Float16* __restrict__ WxT,
                                                    const float* __restrict__ bg,
                                                    _Float16* __restrict__ Xp) {
  extern __shared__ char smem[];
  _Float16* ldsA = (_Float16*)smem;                    // [128][264] (pad 8)
  _Float16* ldsB = (_Float16*)(smem + 128 * 264 * 2);  // [128][264]
  const int mtile = blockIdx.x;   // 0..255
  const int ntile = blockIdx.y;   // 0..15
  const int tid = threadIdx.x;

  // Load B tile: WxT rows n = ntile*128 .. +127, all 256 k
#if USE_ASYNC_LDS
  for (int idx = tid; idx < 128 * 32; idx += 256) {   // 16-byte chunks
    int row = idx >> 5, ch = idx & 31;
    async_copy_b128(&WxT[((size_t)(ntile * 128 + row)) * kE + ch * 8],
                    &ldsB[row * 264 + ch * 8]);
  }
#else
  for (int i = 0; i < 128; ++i) {
    ldsB[i * 264 + tid] = WxT[((size_t)(ntile * 128 + i)) * kE + tid];
  }
#endif

  // Gather 128 embedding rows (f32->f16); tid = k (coalesced)
  for (int i = 0; i < 128; ++i) {
    int r = mtile * 128 + i;            // r = t*64 + b
    int t = r >> 6, b = r & 63;
    int tok = x[b * kT + t];
    ldsA[i * 264 + tid] = (_Float16)emb[(size_t)tok * kE + tid];
  }

#if USE_ASYNC_LDS
  __builtin_amdgcn_s_wait_asynccnt(0);
#endif
  __syncthreads();

  const int wave = tid >> 5, lane = tid & 31;
  const int lrow = lane & 15;
  const int klo = (lane < 16) ? 0 : 8;    // A-fragment K base
  const int kbB = (lane < 16) ? 0 : 8;    // B-fragment K base
  const int rofs = (lane < 16) ? 0 : 8;   // D-fragment row offset

  v8f acc[8];
#pragma unroll
  for (int ns = 0; ns < 8; ++ns) acc[ns] = zero8();

  for (int kk = 0; kk < 8; ++kk) {        // K = 256 in steps of 32
    const _Float16* ap = &ldsA[(wave * 16 + lrow) * 264 + kk * 32 + klo];
    v16h afrag = cat16(*(const v8h*)ap, *(const v8h*)(ap + 16));
#pragma unroll
    for (int ns = 0; ns < 8; ++ns) {
      const _Float16* bp = &ldsB[(ns * 16 + lrow) * 264 + kk * 32 + kbB];
      v16h bfrag = cat16(*(const v8h*)bp, *(const v8h*)(bp + 16));
      acc[ns] = __builtin_amdgcn_wmma_f32_16x16x32_f16(false, afrag, false, bfrag,
                                                       (short)0, acc[ns], false, false);
    }
  }

  const int colbase = ntile * 128;
#pragma unroll
  for (int ns = 0; ns < 8; ++ns) {
    int col = colbase + ns * 16 + lrow;
    float bgv = bg[col];
#pragma unroll
    for (int r = 0; r < 8; ++r) {
      int row = wave * 16 + r + rofs;
      Xp[((size_t)(mtile * 128 + row)) * kG + col] = (_Float16)(acc[ns][r] + bgv);
    }
  }
}

// ---------------------------------------------------------------------------
// Kernel 2: persistent LSTM recurrence. 16 WGs, WG g owns hidden j in
// [g*32, g*32+32) -> 128 gate cols {i,f,g,o}. Wh slice lives in LDS all 512
// steps. h ping-pongs in global f16; step barrier via atomic counters.
// ---------------------------------------------------------------------------
__global__ __launch_bounds__(256) void lstm_rec(const _Float16* __restrict__ WhT,
                                                const _Float16* __restrict__ Xp,
                                                _Float16* __restrict__ h0buf,
                                                _Float16* __restrict__ h1buf,
                                                int* __restrict__ cnt) {
  extern __shared__ char smem[];
  _Float16* WhS = (_Float16*)smem;                                  // [128][528]
  float* gates  = (float*)(smem + 128 * 528 * 2);                   // [64][132]
  float* cbuf   = (float*)(smem + 128 * 528 * 2 + 64 * 132 * 4);    // [64][32]
  const int g = blockIdx.x;
  const int j0 = g * 32;
  const int tid = threadIdx.x;

  // Preload Wh slice: local col cl -> global gate col (cl/32)*512 + j0 + cl%32
#if USE_ASYNC_LDS
  for (int idx = tid; idx < 128 * 64; idx += 256) {   // 16-byte chunks
    int cl = idx >> 6, ch = idx & 63;
    int n = (cl >> 5) * kH + j0 + (cl & 31);
    async_copy_b128(&WhT[(size_t)n * kH + ch * 8], &WhS[cl * 528 + ch * 8]);
  }
#else
  for (int idx = tid; idx < 128 * 512; idx += 256) {
    int cl = idx >> 9, k = idx & 511;
    int n = (cl >> 5) * kH + j0 + (cl & 31);
    WhS[cl * 528 + k] = WhT[(size_t)n * kH + k];
  }
#endif
  for (int idx = tid; idx < 64 * 32; idx += 256) cbuf[idx] = 0.0f;
#if USE_ASYNC_LDS
  __builtin_amdgcn_s_wait_asynccnt(0);
#endif
  __syncthreads();

  const int wave = tid >> 5, lane = tid & 31;
  const int lrow = lane & 15;
  const int klo = (lane < 16) ? 0 : 8;
  const int kbB = (lane < 16) ? 0 : 8;
  const int rofs = (lane < 16) ? 0 : 8;
  const int m_sub = wave & 3;            // 16-row batch strip
  const int nbase = (wave >> 2) * 4;     // 4 of 8 N-subtiles

  _Float16* hb[2] = {h0buf, h1buf};
  const int ejj = tid & 31, eb0 = tid >> 5;

  for (int t = 0; t < kT; ++t) {
    const _Float16* hr = hb[t & 1];
    _Float16* hw = hb[(t + 1) & 1];

    // Prefetch next step's Xproj slice (global_prefetch_b8)
    if (t + 1 < kT) {
      int pb = tid >> 2, pq = tid & 3;
      __builtin_prefetch(&Xp[((size_t)(t + 1) * kB + pb) * kG + pq * kH + j0], 0, 1);
    }

    // gates_partial = h @ Wh_slice : (64x512)@(512x128) via WMMA
    v8f acc[4];
#pragma unroll
    for (int ni = 0; ni < 4; ++ni) acc[ni] = zero8();

    for (int kk = 0; kk < 16; ++kk) {    // K = 512 in steps of 32
      const _Float16* ap = &hr[(m_sub * 16 + lrow) * kH + kk * 32 + klo];
      v16h afrag = cat16(*(const v8h*)ap, *(const v8h*)(ap + 16));
#pragma unroll
      for (int ni = 0; ni < 4; ++ni) {
        const _Float16* bp = &WhS[((nbase + ni) * 16 + lrow) * 528 + kk * 32 + kbB];
        v16h bfrag = cat16(*(const v8h*)bp, *(const v8h*)(bp + 16));
        acc[ni] = __builtin_amdgcn_wmma_f32_16x16x32_f16(false, afrag, false, bfrag,
                                                         (short)0, acc[ni], false, false);
      }
    }
#pragma unroll
    for (int ni = 0; ni < 4; ++ni) {
      int gcol = (nbase + ni) * 16 + lrow;
#pragma unroll
      for (int r = 0; r < 8; ++r) {
        int grow = m_sub * 16 + r + rofs;
        gates[grow * 132 + gcol] = acc[ni][r];
      }
    }
    __syncthreads();

    // Elementwise LSTM cell update for our 32 hidden units x 64 batch
    for (int i = 0; i < 8; ++i) {
      int b = eb0 + i * 8;
      const _Float16* xp = &Xp[((size_t)t * kB + b) * kG + j0 + ejj];
      float iv = gates[b * 132 +       ejj] + (float)xp[0];
      float fv = gates[b * 132 + 32  + ejj] + (float)xp[512];
      float gv = gates[b * 132 + 64  + ejj] + (float)xp[1024];
      float ov = gates[b * 132 + 96  + ejj] + (float)xp[1536];
      float c = sigf(fv) * cbuf[b * 32 + ejj] + sigf(iv) * tanhf(gv);
      cbuf[b * 32 + ejj] = c;
      float h = sigf(ov) * tanhf(c);
      hw[b * kH + j0 + ejj] = (_Float16)h;
    }

    // Device-wide step barrier (release writes, arrive, spin with acquire)
    __threadfence();
    __syncthreads();
    if (tid == 0)
      __hip_atomic_fetch_add(&cnt[t], 1, __ATOMIC_ACQ_REL, __HIP_MEMORY_SCOPE_AGENT);
    while (__hip_atomic_load(&cnt[t], __ATOMIC_ACQUIRE, __HIP_MEMORY_SCOPE_AGENT) < kNWG) {
      __builtin_amdgcn_s_sleep(1);
    }
  }
}

// ---------------------------------------------------------------------------
// Kernel 3: out[b][c] = h @ Vw + Vb   (64x512 @ 512x2, tiny)
// ---------------------------------------------------------------------------
__global__ __launch_bounds__(128) void head_kernel(const _Float16* __restrict__ hfin,
                                                   const float* __restrict__ Vw,
                                                   const float* __restrict__ Vb,
                                                   float* __restrict__ out) {
  int tid = threadIdx.x;       // 128 = 64 batch * 2 classes
  int b = tid >> 1, cl = tid & 1;
  float acc = Vb[cl];
  for (int j = 0; j < kH; ++j)
    acc += (float)hfin[b * kH + j] * Vw[j * 2 + cl];
  out[b * 2 + cl] = acc;
}

extern "C" void kernel_launch(void* const* d_in, const int* in_sizes, int n_in,
                              void* d_out, int out_size, void* d_ws, size_t ws_size,
                              hipStream_t stream) {
  (void)in_sizes; (void)n_in; (void)out_size; (void)ws_size;
  const int*   x   = (const int*)d_in[0];
  const float* emb = (const float*)d_in[1];
  const float* Wg  = (const float*)d_in[2];
  const float* bg  = (const float*)d_in[3];
  const float* Vw  = (const float*)d_in[4];
  const float* Vb  = (const float*)d_in[5];
  float* out = (float*)d_out;

  char* ws = (char*)d_ws;
  _Float16* WxT = (_Float16*)(ws + WXT_OFF);
  _Float16* WhT = (_Float16*)(ws + WHT_OFF);
  _Float16* h0  = (_Float16*)(ws + H0_OFF);
  _Float16* h1  = (_Float16*)(ws + H1_OFF);
  int*      cnt = (int*)(ws + CNT_OFF);
  _Float16* Xp  = (_Float16*)(ws + XP_OFF);

  // Zero h ping-pong buffers + barrier counters (each launch; graph-safe)
  (void)hipMemsetAsync(ws + H0_OFF, 0, 128 * 1024 + kT * sizeof(int), stream);

  // 0) weight convert+transpose
  cvt_weights<<<(768 * 2048) / 256, 256, 0, stream>>>(Wg, WxT, WhT);

  // 1) Xproj = embed-gather @ Wx + bg  (WMMA GEMM, fully parallel)
  xproj_kernel<<<dim3(256, 16), 256, 2 * 128 * 264 * 2, stream>>>(x, emb, WxT, bg, Xp);

  // 2) persistent recurrence (16 WGs, Wh slice resident in 320KB LDS)
  size_t rec_lds = 128 * 528 * 2 + 64 * 132 * 4 + 64 * 32 * 4;
  lstm_rec<<<kNWG, 256, rec_lds, stream>>>(WhT, Xp, h0, h1, cnt);

  // 3) classifier head; final h is in buffer (T % 2) == 0
  head_kernel<<<1, 128, 0, stream>>>(h0, Vw, Vb, out);
}
